// rnn_decoder_37993280700569
// MI455X (gfx1250) — matrix operand order repaired and checked by reading, not compiled, not run on hardware
//
#include <hip/hip_runtime.h>
#include <hip/hip_bf16.h>

// ---------------- problem constants ----------------
#define BB   16
#define HH   32
#define WWD  64
#define HW   2048            // 32*64
#define SP   (BB*HW)         // 32768 spatial rows
#define HIDN 256
#define ENCC 1024
#define VOC  5000

typedef __attribute__((ext_vector_type(16))) _Float16 v16h;
typedef __attribute__((ext_vector_type(8)))  _Float16 v8h;
typedef __attribute__((ext_vector_type(8)))  float    v8f;
typedef __attribute__((ext_vector_type(4)))  unsigned u32x4;
typedef __attribute__((ext_vector_type(8)))  int      i32x8;
typedef __attribute__((ext_vector_type(4)))  int      i32x4;

// =====================================================================
// Tensor Data Mover: contiguous 1-D tile, global -> LDS
// D# per CDNA5 ISA (data_size=3 -> 8-byte units)
// =====================================================================
__device__ __forceinline__ void tdm_load_1d(unsigned lds_off, const void* gptr,
                                            unsigned bytes) {
    unsigned long long ga = (unsigned long long)gptr;
    unsigned units = bytes >> 3;                       // 8-byte elements
    u32x4 g0;
    g0[0] = 1u;                                        // count=1, user mode
    g0[1] = lds_off;                                   // lds_addr (bytes)
    g0[2] = (unsigned)ga;                              // global_addr[31:0]
    g0[3] = ((unsigned)(ga >> 32) & 0x1FFFFFFu) | (2u << 30);  // addr[56:32] | type=2
    i32x8 g1;
    g1[0] = (int)(3u << 16);                           // data_size=3 (8B)
    g1[1] = (int)((units & 0xFFFFu) << 16);            // tensor_dim0 lo16
    g1[2] = (int)((units >> 16) | (1u << 16));         // tensor_dim0 hi | tensor_dim1=1
    g1[3] = (int)((units & 0xFFFFu) << 16);            // tile_dim0 = units
    g1[4] = 0;                                         // tile_dim1/2 unused
    g1[5] = (int)units;                                // tensor_dim0_stride lo32
    g1[6] = 0;
    g1[7] = 0;
    i32x4 g2 = {0, 0, 0, 0};
    i32x4 g3 = {0, 0, 0, 0};
#if __clang_major__ >= 23
    i32x8 g4 = {0, 0, 0, 0, 0, 0, 0, 0};
    __builtin_amdgcn_tensor_load_to_lds(g0, g1, g2, g3, g4, 0);
#else
    __builtin_amdgcn_tensor_load_to_lds(g0, g1, g2, g3, 0);
#endif
}

// =====================================================================
// small helper kernels (negligible FLOPs)
// =====================================================================

__global__ void k_emb(const int* __restrict__ ia, const float* __restrict__ emb_w,
                      float* __restrict__ emb) {
    int idx = blockIdx.x * blockDim.x + threadIdx.x;
    if (idx >= BB * 256) return;
    int b = idx >> 8, e = idx & 255;
    emb[idx] = emb_w[(size_t)ia[b] * 256 + e];
}

// C[m][n] = (acc? C : 0) + bias[n] + sum_k A[m][k] * Bw[n][k]
__global__ void k_small_gemm(float* __restrict__ C, const float* __restrict__ A,
                             const float* __restrict__ Bw, const float* __restrict__ bias,
                             int M, int N, int K, int accum) {
    int idx = blockIdx.x * blockDim.x + threadIdx.x;
    if (idx >= M * N) return;
    int m = idx / N, n = idx % N;
    float s = bias ? bias[n] : 0.0f;
    const float* a = A + (size_t)m * K;
    const float* b = Bw + (size_t)n * K;
    for (int k = 0; k < K; ++k) s += a[k] * b[k];
    if (accum) C[idx] += s; else C[idx] = s;
}

__global__ void k_gru(const float* __restrict__ gi, const float* __restrict__ gh,
                      const float* __restrict__ hprev, float* __restrict__ hout) {
    int idx = blockIdx.x * blockDim.x + threadIdx.x;
    if (idx >= BB * HIDN) return;
    int b = idx >> 8, h = idx & 255;
    const float* gib = gi + (size_t)b * 768;
    const float* ghb = gh + (size_t)b * 768;
    float r = 1.0f / (1.0f + expf(-(gib[h]       + ghb[h])));
    float z = 1.0f / (1.0f + expf(-(gib[256 + h] + ghb[256 + h])));
    float n = tanhf(gib[512 + h] + r * ghb[512 + h]);
    hout[idx] = (1.0f - z) * n + z * hprev[idx];
}

// att_sum = attention_sum + conv3x3(decoder_attention, conv1_w) + conv1_b
__global__ void k_attsum(const float* __restrict__ att_in, const float* __restrict__ dec_att,
                         const float* __restrict__ cw, const float* __restrict__ cb,
                         float* __restrict__ att_out) {
    int idx = blockIdx.x * blockDim.x + threadIdx.x;
    if (idx >= SP) return;
    int b = idx >> 11, s = idx & 2047, y = s >> 6, x = s & 63;
    float acc = cb[0];
    for (int dy = 0; dy < 3; ++dy)
        for (int dx = 0; dx < 3; ++dx) {
            int yy = y + dy - 1, xx = x + dx - 1;
            if (yy >= 0 && yy < HH && xx >= 0 && xx < WWD)
                acc += dec_att[(size_t)b * HW + yy * WWD + xx] * cw[dy * 3 + dx];
        }
    att_out[idx] = att_in[idx] + acc;
}

// encoder NCHW f32 -> NHWC f16 (unit-stride K for the WMMA A operand)
__global__ void k_cvt_enc(const float* __restrict__ enc, _Float16* __restrict__ ench) {
    int bc = blockIdx.x;                 // b*1024 + c
    int b = bc >> 10, c = bc & 1023;
    const float* src = enc + (size_t)bc * HW;
    for (int s = threadIdx.x; s < HW; s += 256)
        ench[((size_t)b * HW + s) * ENCC + c] = (_Float16)src[s];
}

// ua_w [256][1024] f32 -> fragment-major f16: [ntile16][kblk32][lane32][e16]
// B frag: lane holds N = n0+(lane&15), elem e -> K = kblk*32 + (e<8 ? 8*(lane>>4)+e : 16+8*(lane>>4)+e-8)
__global__ void k_pack_ua(const float* __restrict__ ua_w, _Float16* __restrict__ uaP) {
    int idx = blockIdx.x * blockDim.x + threadIdx.x;
    if (idx >= HIDN * ENCC) return;
    int e     = idx & 15;
    int lane  = (idx >> 4) & 31;
    int kblk  = (idx >> 9) & 31;
    int ntile = idx >> 14;
    int n = ntile * 16 + (lane & 15);
    int k = kblk * 32 + (e < 8 ? 8 * (lane >> 4) + e : 16 + 8 * (lane >> 4) + (e - 8));
    uaP[idx] = (_Float16)ua_w[(size_t)n * ENCC + k];
}

// convtan_w OIHW (co,ci,3,3) -> fragment-major f16: [ntile16][kp9][kblk8][lane32][e16]
__global__ void k_pack_wconv(const float* __restrict__ w, _Float16* __restrict__ wP) {
    int idx = blockIdx.x * blockDim.x + threadIdx.x;
    if (idx >= 16 * 9 * 8 * 32 * 16) return;
    int e    = idx & 15;
    int lane = (idx >> 4) & 31;
    int kb   = (idx >> 9) & 7;
    int t    = idx >> 12;
    int kp   = t % 9;
    int ntile = t / 9;
    int co = ntile * 16 + (lane & 15);
    int ci = kb * 32 + (e < 8 ? 8 * (lane >> 4) + e : 16 + 8 * (lane >> 4) + (e - 8));
    wP[idx] = (_Float16)w[((size_t)co * 256 + ci) * 9 + kp];
}

// e[s] = v_b + sum_n et2[s][n] * v_w[n]
__global__ void k_e_reduce(const float* __restrict__ et2, const float* __restrict__ vw,
                           const float* __restrict__ vb, float* __restrict__ e) {
    int s = blockIdx.x * blockDim.x + threadIdx.x;
    if (s >= SP) return;
    const float* p = et2 + (size_t)s * 256;
    float acc = vb[0];
    for (int n = 0; n < 256; ++n) acc += p[n] * vw[n];
    e[s] = acc;
}

__global__ void k_alpha(const float* __restrict__ e, const int* __restrict__ hm,
                        const int* __restrict__ wm, float* __restrict__ alpha) {
    __shared__ float red[256];
    int b = blockIdx.x, t = threadIdx.x;
    int hmv = hm[b], wmv = wm[b];
    const float* ep = e + (size_t)b * HW;
    float s = 0.0f;
    for (int i = t; i < HW; i += 256) {
        int y = i >> 6, x = i & 63;
        if (y < hmv && x < wmv) s += expf(ep[i]);
    }
    red[t] = s; __syncthreads();
    for (int o = 128; o > 0; o >>= 1) { if (t < o) red[t] += red[t + o]; __syncthreads(); }
    float inv = 1.0f / (red[0] + 1e-8f);
    for (int i = t; i < HW; i += 256) {
        int y = i >> 6, x = i & 63;
        float m = (y < hmv && x < wmv) ? 1.0f : 0.0f;
        alpha[(size_t)b * HW + i] = expf(ep[i]) * m * inv;
    }
}

// ct[b][c] = sum_s alpha[b][s] * enc[b][c][s]
__global__ void k_ct(const float* __restrict__ alpha, const float* __restrict__ enc,
                     float* __restrict__ ct) {
    __shared__ float red[256];
    int bc = blockIdx.x, t = threadIdx.x;
    int b = bc >> 10;
    const float* ap = alpha + (size_t)b * HW;
    const float* ep = enc + (size_t)bc * HW;
    float s = 0.0f;
    for (int i = t; i < HW; i += 256) s += ap[i] * ep[i];
    red[t] = s; __syncthreads();
    for (int o = 128; o > 0; o >>= 1) { if (t < o) red[t] += red[t + o]; __syncthreads(); }
    if (t == 0) ct[bc] = red[0];
}

__global__ void k_lsm(const float* __restrict__ logits, float* __restrict__ out) {
    __shared__ float red[256];
    int b = blockIdx.x, t = threadIdx.x;
    const float* lp = logits + (size_t)b * VOC;
    float mx = -3.4e38f;
    for (int v = t; v < VOC; v += 256) mx = fmaxf(mx, lp[v]);
    red[t] = mx; __syncthreads();
    for (int o = 128; o > 0; o >>= 1) { if (t < o) red[t] = fmaxf(red[t], red[t + o]); __syncthreads(); }
    float M = red[0]; __syncthreads();
    float s = 0.0f;
    for (int v = t; v < VOC; v += 256) s += expf(lp[v] - M);
    red[t] = s; __syncthreads();
    for (int o = 128; o > 0; o >>= 1) { if (t < o) red[t] += red[t + o]; __syncthreads(); }
    float lse = M + logf(red[0]);
    for (int v = t; v < VOC; v += 256) out[(size_t)b * VOC + v] = lp[v] - lse;
}

// =====================================================================
// WMMA kernels (v_wmma_f32_16x16x32_f16)
// A f16 16x32: lane<16 -> K 0..7 / 16..23; lane>=16 -> K 8..15 / 24..31
// B f16 32x16: lane holds N=lane%16, elems = K in the same order as A
// C/D f32 16x16: elem r -> row r + 8*(lane>=16), col lane%16
// B tiles are staged to LDS by the Tensor Data Mover in fragment-major order.
// =====================================================================

// et = enc_nhwc @ ua_w^T + ua_b + hidden1 + att_sum*uf_w + uf_b   (f16 out)
__global__ __launch_bounds__(256)
void k_et_wmma(const _Float16* __restrict__ enc_h,   // [SP][1024] NHWC f16
               const _Float16* __restrict__ uaP,     // [16][32][32][16] packed
               const float* __restrict__ ua_b, const float* __restrict__ uf_w,
               const float* __restrict__ uf_b, const float* __restrict__ hid1,
               const float* __restrict__ att_sum,    // [SP]
               _Float16* __restrict__ et)            // [SP][256] f16
{
    __shared__ _Float16 btile[32 * 512];             // 32 KB, one N-tile of B

    const int ntile = blockIdx.x & 15;
    const int mtile = (blockIdx.x >> 4) * 8 + (threadIdx.x >> 5);
    const int lane = threadIdx.x & 31;
    const int n0 = ntile * 16;
    const int row0 = mtile * 16;
    const int mrow = lane & 15;
    const int half = lane >> 4;

    if (threadIdx.x < 32) {                           // wave 0 issues the TDM
        tdm_load_1d((unsigned)(unsigned long long)(void*)btile,
                    uaP + (size_t)ntile * (32 * 512), 32 * 512 * 2);
        __builtin_amdgcn_s_wait_tensorcnt(0);
    }
    __syncthreads();

    const _Float16* arow = enc_h + (size_t)(row0 + mrow) * ENCC;
    const _Float16* bl = btile + lane * 16;

    v8f acc = {};
    for (int kb = 0; kb < 32; ++kb) {
        const int k0 = kb * 32;
        v8h alo = *(const v8h*)(arow + k0 + 8 * half);
        v8h ahi = *(const v8h*)(arow + k0 + 16 + 8 * half);
        v16h a;
#pragma unroll
        for (int e = 0; e < 8; ++e) { a[e] = alo[e]; a[e + 8] = ahi[e]; }
        v16h b = *(const v16h*)(bl + kb * 512);
        acc = __builtin_amdgcn_wmma_f32_16x16x32_f16(false, a, false, b,
                                                     (short)0, acc, false, false);
    }

    const int bimg = row0 >> 11;
    const int n = n0 + mrow;
    const float hb  = hid1[bimg * 256 + n] + ua_b[n] + uf_b[n];
    const float ufn = uf_w[n];
#pragma unroll
    for (int r = 0; r < 8; ++r) {
        int sr = row0 + r + 8 * half;
        et[(size_t)sr * 256 + n] = (_Float16)(acc[r] + hb + att_sum[sr] * ufn);
    }
}

// et2 = tanh(BN(mask * (conv3x3(et, convtan_w) + convtan_b)))   (implicit GEMM)
__global__ __launch_bounds__(256)
void k_conv_wmma(const _Float16* __restrict__ et,    // [SP][256] f16 NHWC
                 const _Float16* __restrict__ wP,    // [16][9][8][32][16] packed
                 const float* __restrict__ conv_b,
                 const int* __restrict__ h_mask, const int* __restrict__ w_mask,
                 const float* __restrict__ bn_g, const float* __restrict__ bn_b,
                 const float* __restrict__ bn_rm, const float* __restrict__ bn_rv,
                 float* __restrict__ et2)
{
    __shared__ _Float16 wtile[9 * 8 * 512];          // 72 KB, one N-tile of weights

    const int ntile = blockIdx.x & 15;
    const int mtile = (blockIdx.x >> 4) * 8 + (threadIdx.x >> 5);
    const int lane = threadIdx.x & 31;
    const int n0 = ntile * 16;
    const int row0 = mtile * 16;
    const int mrow = lane & 15;
    const int half = lane >> 4;

    if (threadIdx.x < 32) {
        tdm_load_1d((unsigned)(unsigned long long)(void*)wtile,
                    wP + (size_t)ntile * (9 * 8 * 512), 9 * 8 * 512 * 2);
        __builtin_amdgcn_s_wait_tensorcnt(0);
    }
    __syncthreads();

    const int bimg = row0 >> 11;
    const int s = row0 & 2047;
    const int y = s >> 6;
    const int x0 = s & 63;

    v8h vz = {};
    v8f acc = {};
    for (int ky = 0; ky < 3; ++ky) {
        int yy = y + ky - 1;
        if (yy < 0 || yy >= HH) continue;            // wave-uniform branch
        for (int kx = 0; kx < 3; ++kx) {
            int xx = x0 + mrow + kx - 1;
            bool ok = (xx >= 0) && (xx < WWD);
            const _Float16* ap = et + ((size_t)bimg * HW + yy * WWD + xx) * 256;
            const _Float16* wl = wtile + (size_t)(ky * 3 + kx) * (8 * 512) + lane * 16;
            for (int kb = 0; kb < 8; ++kb) {
                const int k0 = kb * 32;
                v8h alo = ok ? *(const v8h*)(ap + k0 + 8 * half) : vz;
                v8h ahi = ok ? *(const v8h*)(ap + k0 + 16 + 8 * half) : vz;
                v16h a;
#pragma unroll
                for (int e = 0; e < 8; ++e) { a[e] = alo[e]; a[e + 8] = ahi[e]; }
                v16h b = *(const v16h*)(wl + kb * 512);
                acc = __builtin_amdgcn_wmma_f32_16x16x32_f16(false, a, false, b,
                                                             (short)0, acc, false, false);
            }
        }
    }

    const int n = n0 + mrow;
    const float g   = bn_g[n] * rsqrtf(bn_rv[n] + 1e-5f);
    const float rm  = bn_rm[n];
    const float bnb = bn_b[n];
    const float cb  = conv_b[n];
    const int hmv = h_mask[bimg], wmv = w_mask[bimg];
    const float ym = (y < hmv) ? 1.0f : 0.0f;
#pragma unroll
    for (int r = 0; r < 8; ++r) {
        int xr = x0 + r + 8 * half;
        float m = ym * ((xr < wmv) ? 1.0f : 0.0f);
        float v = (acc[r] + cb) * m;
        v = (v - rm) * g + bnb;
        et2[((size_t)bimg * HW + y * WWD + xr) * 256 + n] = tanhf(v);
    }
}

// logits = pre @ out_w^T + out_b    (M=16, N=5000, K=128)
__global__ __launch_bounds__(256)
void k_vocab_wmma(const float* __restrict__ pre,     // [16][128]
                  const float* __restrict__ out_w,   // [5000][128]
                  const float* __restrict__ out_b,
                  float* __restrict__ logits)        // [16][5000]
{
    const int wave = (blockIdx.x * blockDim.x + threadIdx.x) >> 5;
    const int NT = (VOC + 15) / 16;                  // 313 tiles
    if (wave >= NT) return;                          // whole-wave exit
    const int lane = threadIdx.x & 31;
    const int n0 = wave * 16;
    const int mrow = lane & 15;
    const int half = lane >> 4;

    const int n = n0 + mrow;
    const bool nok = n < VOC;
    const float* ap = pre + (size_t)mrow * 128;
    const float* bp = out_w + (size_t)(nok ? n : 0) * 128;

    v8f acc = {};
    for (int k0 = 0; k0 < 128; k0 += 32) {
        v16h a, b;
#pragma unroll
        for (int e = 0; e < 8; ++e) {
            a[e]     = (_Float16)ap[k0 + 8 * half + e];
            a[e + 8] = (_Float16)ap[k0 + 16 + 8 * half + e];
        }
#pragma unroll
        for (int e = 0; e < 16; ++e)
            b[e] = nok ? (_Float16)bp[k0 + (e < 8 ? 8 * half + e : 16 + 8 * half + (e - 8))] : (_Float16)0.0f;
        acc = __builtin_amdgcn_wmma_f32_16x16x32_f16(false, a, false, b,
                                                     (short)0, acc, false, false);
    }
    if (nok) {
        float ob = out_b[n];
#pragma unroll
        for (int r = 0; r < 8; ++r) {
            int row = r + 8 * half;                  // row < 16 == B
            logits[(size_t)row * VOC + n] = acc[r] + ob;
        }
    }
}

// =====================================================================
// launch
// =====================================================================
extern "C" void kernel_launch(void* const* d_in, const int* in_sizes, int n_in,
                              void* d_out, int out_size, void* d_ws, size_t ws_size,
                              hipStream_t stream) {
    (void)in_sizes; (void)n_in; (void)out_size; (void)ws_size;

    const int*   input_a   = (const int*)  d_in[0];
    const float* in_hidden = (const float*)d_in[1];
    const float* enc       = (const float*)d_in[2];
    const float* att_in    = (const float*)d_in[3];
    const float* dec_att   = (const float*)d_in[4];
    const int*   h_mask    = (const int*)  d_in[5];
    const int*   w_mask    = (const int*)  d_in[6];
    const float* emb_w     = (const float*)d_in[11];
    const float* g1_wih    = (const float*)d_in[12];
    const float* g1_whh    = (const float*)d_in[13];
    const float* g1_bih    = (const float*)d_in[14];
    const float* g1_bhh    = (const float*)d_in[15];
    const float* fc1_w     = (const float*)d_in[16];
    const float* fc1_b     = (const float*)d_in[17];
    const float* g2_wih    = (const float*)d_in[18];
    const float* g2_whh    = (const float*)d_in[19];
    const float* g2_bih    = (const float*)d_in[20];
    const float* g2_bhh    = (const float*)d_in[21];
    const float* out_w     = (const float*)d_in[22];
    const float* out_b     = (const float*)d_in[23];
    const float* emb2_w    = (const float*)d_in[24];
    const float* emb2_b    = (const float*)d_in[25];
    const float* conv1_w   = (const float*)d_in[26];
    const float* conv1_b   = (const float*)d_in[27];
    const float* convt_w   = (const float*)d_in[28];
    const float* convt_b   = (const float*)d_in[29];
    const float* fc2_w     = (const float*)d_in[30];
    const float* fc2_b     = (const float*)d_in[31];
    const float* ua_w      = (const float*)d_in[32];
    const float* ua_b      = (const float*)d_in[33];
    const float* uf_w      = (const float*)d_in[34];
    const float* uf_b      = (const float*)d_in[35];
    const float* v_w       = (const float*)d_in[36];
    const float* v_b       = (const float*)d_in[37];
    const float* wc_w      = (const float*)d_in[38];
    const float* wc_b      = (const float*)d_in[39];
    const float* bn_g      = (const float*)d_in[40];
    const float* bn_b      = (const float*)d_in[41];
    const float* bn_rm     = (const float*)d_in[42];
    const float* bn_rv     = (const float*)d_in[43];

    float* out = (float*)d_out;
    float* out_logp  = out;                 // 16*5000
    float* out_hid   = out + 80000;         // 16*256
    float* out_alpha = out + 84096;         // 16*2048
    float* out_att   = out + 116864;        // 16*2048

    char* wptr = (char*)d_ws;
    auto alloc = [&](size_t bytes) -> char* {
        char* p = wptr;
        wptr += (bytes + 255) & ~(size_t)255;
        return p;
    };
    _Float16* enc_h = (_Float16*)alloc((size_t)SP * ENCC * 2);      // 64 MB
    _Float16* uaP   = (_Float16*)alloc((size_t)HIDN * ENCC * 2);    // 512 KB
    _Float16* wP    = (_Float16*)alloc((size_t)9 * 256 * 256 * 2);  // 1.1 MB
    _Float16* et    = (_Float16*)alloc((size_t)SP * 256 * 2);       // 16 MB
    float* et2    = (float*)alloc((size_t)SP * 256 * 4);            // 32 MB
    float* emb    = (float*)alloc((size_t)BB * 256 * 4);
    float* gi     = (float*)alloc((size_t)BB * 768 * 4);
    float* gh     = (float*)alloc((size_t)BB * 768 * 4);
    float* st     = (float*)alloc((size_t)BB * 256 * 4);
    float* hid1   = (float*)alloc((size_t)BB * 256 * 4);
    float* ebuf   = (float*)alloc((size_t)SP * 4);
    float* ct     = (float*)alloc((size_t)BB * ENCC * 4);
    float* gi2    = (float*)alloc((size_t)BB * 768 * 4);
    float* gh2    = (float*)alloc((size_t)BB * 768 * 4);
    float* pre    = (float*)alloc((size_t)BB * 128 * 4);
    float* logits = (float*)alloc((size_t)BB * VOC * 4);

    // precision conversion / fragment packing for WMMA operands
    k_cvt_enc   <<<BB * ENCC, 256, 0, stream>>>(enc, enc_h);
    k_pack_ua   <<<(HIDN * ENCC + 255) / 256, 256, 0, stream>>>(ua_w, uaP);
    k_pack_wconv<<<(16 * 9 * 8 * 32 * 16 + 255) / 256, 256, 0, stream>>>(convt_w, wP);

    // GRU1 + fc1
    k_emb<<<(BB * 256 + 255) / 256, 256, 0, stream>>>(input_a, emb_w, emb);
    k_small_gemm<<<(BB * 768 + 255) / 256, 256, 0, stream>>>(gi, emb, g1_wih, g1_bih, BB, 768, 256, 0);
    k_small_gemm<<<(BB * 768 + 255) / 256, 256, 0, stream>>>(gh, in_hidden, g1_whh, g1_bhh, BB, 768, 256, 0);
    k_gru<<<(BB * 256 + 255) / 256, 256, 0, stream>>>(gi, gh, in_hidden, st);
    k_small_gemm<<<(BB * 256 + 255) / 256, 256, 0, stream>>>(hid1, st, fc1_w, fc1_b, BB, 256, 256, 0);

    // coverage attention sum (output #4)
    k_attsum<<<(SP + 255) / 256, 256, 0, stream>>>(att_in, dec_att, conv1_w, conv1_b, out_att);

    // big WMMA GEMMs (B tiles staged via Tensor Data Mover)
    k_et_wmma  <<<4096, 256, 0, stream>>>(enc_h, uaP, ua_b, uf_w, uf_b, hid1, out_att, et);
    k_conv_wmma<<<4096, 256, 0, stream>>>(et, wP, convt_b, h_mask, w_mask,
                                          bn_g, bn_b, bn_rm, bn_rv, et2);

    // attention scores -> alpha (output #3) -> context ct
    k_e_reduce<<<(SP + 255) / 256, 256, 0, stream>>>(et2, v_w, v_b, ebuf);
    k_alpha<<<BB, 256, 0, stream>>>(ebuf, h_mask, w_mask, out_alpha);
    k_ct<<<BB * ENCC, 256, 0, stream>>>(out_alpha, enc, ct);

    // GRU2 (hidden output #2 written in place)
    k_small_gemm<<<(BB * 768 + 255) / 256, 256, 0, stream>>>(gi2, ct, g2_wih, g2_bih, BB, 768, 1024, 0);
    k_small_gemm<<<(BB * 768 + 255) / 256, 256, 0, stream>>>(gh2, st, g2_whh, g2_bhh, BB, 768, 256, 0);
    k_gru<<<(BB * 256 + 255) / 256, 256, 0, stream>>>(gi2, gh2, st, out_hid);

    // pre = h2@fc2^T + emb@emb2^T + ct@wc^T (+biases)
    k_small_gemm<<<(BB * 128 + 255) / 256, 256, 0, stream>>>(pre, out_hid, fc2_w, fc2_b, BB, 128, 256, 0);
    k_small_gemm<<<(BB * 128 + 255) / 256, 256, 0, stream>>>(pre, emb, emb2_w, emb2_b, BB, 128, 256, 1);
    k_small_gemm<<<(BB * 128 + 255) / 256, 256, 0, stream>>>(pre, ct, wc_w, wc_b, BB, 128, 1024, 1);

    // vocab projection + log-softmax (output #1)
    k_vocab_wmma<<<40, 256, 0, stream>>>(pre, out_w, out_b, logits);
    k_lsm<<<BB, 256, 0, stream>>>(logits, out_logp);
}
